// GATv2WithGlobal_4389456577269
// MI455X (gfx1250) — matrix operand order repaired and checked by reading, not compiled, not run on hardware
//
#include <hip/hip_runtime.h>
#include <hip/hip_bf16.h>
#include <math.h>
#include <stdint.h>

#define N_NODES 131072
#define E_EDGES 262144
#define G_GRAPHS 4096
#define F_INPUT 9
#define HEADS 4
#define C1 64
#define C2 128
#define D1 256          // HEADS*C1
#define D2 512          // HEADS*C2
#define GFEAT 177
#define NEG_SLOPE 0.2f
#define BN_EPS 1e-5f

#define LDS_KSTRIDE 136   // 128 k-elems + 8 pad (bf16) -> lanes hit banks 4 apart

typedef __attribute__((ext_vector_type(16))) __bf16 v16bf;
typedef __attribute__((ext_vector_type(8)))  float  v8f;

// ---- CDNA5 async global->LDS path (probe-confirmed present; typed per clang) ----
#if __has_builtin(__builtin_amdgcn_global_load_async_to_lds_b128) && \
    __has_builtin(__builtin_amdgcn_s_wait_asynccnt)
#define USE_ASYNC_LDS 1
typedef int v4i_ __attribute__((__vector_size__(16)));
typedef __attribute__((address_space(1))) v4i_* async_gptr_t;   // global int4*
typedef __attribute__((address_space(3))) v4i_* async_lptr_t;   // LDS int4*
#else
#define USE_ASYNC_LDS 0
#endif

__device__ inline void g2l_b128(void* ldst, const void* gsrc) {
#if USE_ASYNC_LDS
    __builtin_amdgcn_global_load_async_to_lds_b128(
        (async_gptr_t)(uintptr_t)gsrc,
        (async_lptr_t)(unsigned int)(uintptr_t)ldst,   // low 32 bits = LDS offset
        0, 0);
#else
    *(uint4*)ldst = *(const uint4*)gsrc;
#endif
}

__device__ inline void g2l_wait() {
#if USE_ASYNC_LDS
    __builtin_amdgcn_s_wait_asynccnt(0);
#endif
}

// ---------------------------------------------------------------- utilities

__global__ void fill_f32(float* __restrict__ p, float v, int n) {
    int tid = blockIdx.x * blockDim.x + threadIdx.x;
    if (tid < n) p[tid] = v;
}

__device__ inline void atomicMaxF(float* addr, float val) {
    // signed-max for non-negative floats, unsigned-min for negative floats.
    // Valid when the cell is initialized to -inf (0xFF800000).
    if (val >= 0.0f) atomicMax((int*)addr, __float_as_int(val));
    else             atomicMin((unsigned int*)addr, __float_as_uint(val));
}

// --------------------------------------------- layer 1: K=9 node transforms

__global__ void l1_transform(const float* __restrict__ x,
                             const float* __restrict__ Wl, const float* __restrict__ bl,
                             const float* __restrict__ Wr, const float* __restrict__ br,
                             float* __restrict__ xl, float* __restrict__ xr) {
    int tid = blockIdx.x * blockDim.x + threadIdx.x;       // over N*D1
    if (tid >= N_NODES * D1) return;
    int node = tid / D1;
    int j    = tid - node * D1;
    const float* xi = x + (size_t)node * F_INPUT;
    float sl = bl[j], sr = br[j];
#pragma unroll
    for (int k = 0; k < F_INPUT; ++k) {
        float v = xi[k];
        sl += v * Wl[k * D1 + j];
        sr += v * Wr[k * D1 + j];
    }
    xl[tid] = sl;
    xr[tid] = sr;
}

// ------------------------------------------------------- edge attention ops
// Wave per (edge, head): coalesced channel loads + wave32 shuffle reduction.

template<int C, int D>
__global__ void edge_score_wave(const float* __restrict__ xl, const float* __restrict__ xr,
                                const float* __restrict__ att,
                                const int* __restrict__ srcIdx, const int* __restrict__ dstIdx,
                                float* __restrict__ score, float* __restrict__ mbuf) {
    const int EE = E_EDGES + N_NODES;
    int gw   = (blockIdx.x * blockDim.x + threadIdx.x) >> 5;   // global wave id
    int lane = threadIdx.x & 31;
    if (gw >= EE * HEADS) return;
    int e = gw >> 2;
    int h = gw & 3;
    int s, d;
    if (e < E_EDGES) { s = srcIdx[e]; d = dstIdx[e]; }
    else             { s = d = e - E_EDGES; }                  // self loop
    const float* pl = xl + (size_t)s * D + h * C;
    const float* pr = xr + (size_t)d * D + h * C;
    const float* pa = att + h * C;
    float sc = 0.0f;
#pragma unroll
    for (int c = lane; c < C; c += 32) {
        float v = pl[c] + pr[c];
        v = v > 0.0f ? v : NEG_SLOPE * v;                      // leaky_relu
        sc += v * pa[c];
    }
#pragma unroll
    for (int o = 16; o > 0; o >>= 1) sc += __shfl_xor(sc, o, 32);
    if (lane == 0) {
        score[gw] = sc;
        atomicMaxF(mbuf + (size_t)d * HEADS + h, sc);
    }
}

__global__ void edge_exp(const int* __restrict__ srcIdx, const int* __restrict__ dstIdx,
                         float* __restrict__ score, const float* __restrict__ mbuf,
                         float* __restrict__ denom) {
    const int EE = E_EDGES + N_NODES;
    int tid = blockIdx.x * blockDim.x + threadIdx.x;
    if (tid >= EE * HEADS) return;
    int e = tid >> 2;
    int h = tid & 3;
    int d = (e < E_EDGES) ? dstIdx[e] : (e - E_EDGES);
    float a = __expf(score[tid] - mbuf[(size_t)d * HEADS + h]);
    score[tid] = a;
    atomicAdd(denom + (size_t)d * HEADS + h, a);
}

// Thread per (edge, channel): one coalesced global_atomic_add_f32 per lane.
template<int C, int D>
__global__ void edge_scatter(const float* __restrict__ xl,
                             const int* __restrict__ srcIdx, const int* __restrict__ dstIdx,
                             const float* __restrict__ score, const float* __restrict__ denom,
                             float* __restrict__ agg) {
    const int EE = E_EDGES + N_NODES;
    long long tid = (long long)blockIdx.x * blockDim.x + threadIdx.x;  // over EE*D
    if (tid >= (long long)EE * D) return;
    int e = (int)(tid / D);
    int c = (int)(tid - (long long)e * D);
    int h = c / C;
    int s, d;
    if (e < E_EDGES) { s = srcIdx[e]; d = dstIdx[e]; }
    else             { s = d = e - E_EDGES; }
    float alpha = score[(size_t)e * HEADS + h] / denom[(size_t)d * HEADS + h];
    atomicAdd(agg + (size_t)d * D + c, alpha * xl[(size_t)s * D + c]);
}

// --------------------------------------------------- BN + ReLU (+ bf16 out)

__global__ void bn_relu(const float* __restrict__ agg, const float* __restrict__ bias,
                        const float* __restrict__ gamma, const float* __restrict__ beta,
                        const float* __restrict__ mean,  const float* __restrict__ var,
                        float* __restrict__ outf, __bf16* __restrict__ outb,
                        int D, int total) {
    int tid = blockIdx.x * blockDim.x + threadIdx.x;
    if (tid >= total) return;
    int j = tid % D;
    float v = agg[tid] + bias[j];
    v = (v - mean[j]) * rsqrtf(var[j] + BN_EPS) * gamma[j] + beta[j];
    v = fmaxf(v, 0.0f);
    if (outf) outf[tid] = v;
    if (outb) outb[tid] = (__bf16)v;
}

// ---------------------------------- weight transpose + f32 -> bf16 convert

__global__ void transpose_cvt(const float* __restrict__ W, __bf16* __restrict__ WT,
                              int K, int Nn) {
    int tid = blockIdx.x * blockDim.x + threadIdx.x;       // over K*Nn
    if (tid >= K * Nn) return;
    int k = tid / Nn;
    int n = tid - k * Nn;
    WT[(size_t)n * K + k] = (__bf16)W[tid];
}

// ------------------------------------------------- dual-output bf16 WMMA GEMM
// Computes Cl = A*Wl^T + bl  and  Cr = A*Wr^T + br  in one pass over A.
// A[M x 256] bf16 row-major; WTl/WTr[512 x 256] bf16 N-major.
// Block = 128 threads = 4 waves; wave -> 16(M) x 64(N) strip of BOTH outputs.
// B panels staged in LDS (half-K at a time, both matrices, padded stride).

__global__ void __launch_bounds__(128)
wmma_gemm_dual(const __bf16* __restrict__ A,
               const __bf16* __restrict__ WTl, const __bf16* __restrict__ WTr,
               const float* __restrict__ bl,  const float* __restrict__ br,
               float* __restrict__ Cl,        float* __restrict__ Cr) {
    __shared__ __align__(16) __bf16 ldsB[2][64 * LDS_KSTRIDE];   // ~34 KB

    const int lane  = threadIdx.x & 31;
    const int wave  = threadIdx.x >> 5;
    const int mTile = blockIdx.x * 4 + wave;
    const int nBase = blockIdx.y * 64;
    const int l15   = lane & 15;
    const int hi    = lane >> 4;                 // 0: lanes 0-15, 1: lanes 16-31

    const int rowA = mTile * 16 + l15;
    const __bf16* aRow = A + (size_t)rowA * D1;

    v8f accL[4], accR[4];
#pragma unroll
    for (int t = 0; t < 4; ++t) {
        accL[t] = (v8f){0,0,0,0,0,0,0,0};
        accR[t] = (v8f){0,0,0,0,0,0,0,0};
    }

    for (int half = 0; half < 2; ++half) {
        const int kOff = half * 128;

        // ---- stage B panels: 64 cols x 128 k x 2 matrices (1024 uint4 each)
        for (int i = threadIdx.x; i < 1024; i += 128) {
            const int col = i >> 4;                       // 0..63
            const int kq  = i & 15;                       // 16B chunk within 128 k
            const size_t gsrc = (size_t)(nBase + col) * D1 + kOff + kq * 8;
            g2l_b128(&ldsB[0][col * LDS_KSTRIDE + kq * 8], WTl + gsrc);
            g2l_b128(&ldsB[1][col * LDS_KSTRIDE + kq * 8], WTr + gsrc);
        }
        g2l_wait();
        __syncthreads();

        for (int kk = 0; kk < 128; kk += 32) {
            // ---- A fragment (16-bit A 16x32 layout):
            // lane<16: row l15, K = [k,k+8) u [k+16,k+24); lane>=16: +8 on both.
            union { v16bf v; uint4 u[2]; } af;
            const int abase = kOff + kk + hi * 8;
            af.u[0] = *(const uint4*)(aRow + abase);
            af.u[1] = *(const uint4*)(aRow + abase + 16);
            __builtin_prefetch(aRow + abase + 32, 0, 0);   // global_prefetch_b8

#pragma unroll
            for (int t = 0; t < 4; ++t) {
                // ---- B fragment: col N = nBase+t*16+l15; lanes 0-15 hold
                // K=[kk,kk+8) u [kk+16,kk+24), lanes 16-31 +8 on both (matches A).
                const int boff = (t * 16 + l15) * LDS_KSTRIDE + kk + hi * 8;
                union { v16bf v; uint4 u[2]; } bfu;
                bfu.u[0] = *(const uint4*)(&ldsB[0][boff]);
                bfu.u[1] = *(const uint4*)(&ldsB[0][boff + 16]);
                accL[t] = __builtin_amdgcn_wmma_f32_16x16x32_bf16(
                    false, af.v, false, bfu.v, (short)0, accL[t], false, false);

                bfu.u[0] = *(const uint4*)(&ldsB[1][boff]);
                bfu.u[1] = *(const uint4*)(&ldsB[1][boff + 16]);
                accR[t] = __builtin_amdgcn_wmma_f32_16x16x32_bf16(
                    false, af.v, false, bfu.v, (short)0, accR[t], false, false);
            }
        }
        __syncthreads();
    }

    // ---- store: C/D layout — VGPR i: lanes 0-15 -> M=i, lanes 16-31 -> M=8+i.
#pragma unroll
    for (int t = 0; t < 4; ++t) {
        const int col = nBase + t * 16 + l15;
        const float bL = bl[col];
        const float bR = br[col];
#pragma unroll
        for (int i = 0; i < 8; ++i) {
            const int row = mTile * 16 + i + hi * 8;
            Cl[(size_t)row * D2 + col] = accL[t][i] + bL;
            Cr[(size_t)row * D2 + col] = accR[t][i] + bR;
        }
    }
}

// -------------------------------------------------------------- pooling/head

__global__ void pool_sum(const float* __restrict__ h2, const int* __restrict__ batch,
                         float* __restrict__ pooled) {
    int tid = blockIdx.x * blockDim.x + threadIdx.x;       // over N*D2
    if (tid >= N_NODES * D2) return;
    int node = tid / D2;
    int j    = tid - node * D2;
    atomicAdd(pooled + (size_t)batch[node] * D2 + j, h2[tid]);
}

__global__ void pool_count(const int* __restrict__ batch, float* __restrict__ counts) {
    int tid = blockIdx.x * blockDim.x + threadIdx.x;
    if (tid < N_NODES) atomicAdd(counts + batch[tid], 1.0f);
}

__global__ void head_kernel(const float* __restrict__ pooled, const float* __restrict__ counts,
                            const float* __restrict__ gfeat,
                            const float* __restrict__ fc1W, const float* __restrict__ fc1b,
                            const float* __restrict__ fc2W, const float* __restrict__ fc2b,
                            float* __restrict__ out) {
    __shared__ float z[D2 + GFEAT];
    __shared__ float red[128];
    int g = blockIdx.x;
    int t = threadIdx.x;
    float cnt = fmaxf(counts[g], 1.0f);
    for (int i = t; i < D2; i += 128)    z[i]      = pooled[(size_t)g * D2 + i] / cnt;
    for (int i = t; i < GFEAT; i += 128) z[D2 + i] = gfeat[(size_t)g * GFEAT + i];
    __syncthreads();
    float s = fc1b[t];
    for (int i = 0; i < D2 + GFEAT; ++i) s += z[i] * fc1W[i * 128 + t];
    s = fmaxf(s, 0.0f);
    red[t] = s * fc2W[t];
    __syncthreads();
    for (int off = 64; off > 0; off >>= 1) {
        if (t < off) red[t] += red[t + off];
        __syncthreads();
    }
    if (t == 0) out[g] = red[0] + fc2b[0];
}

// ------------------------------------------------------------------- driver

extern "C" void kernel_launch(void* const* d_in, const int* in_sizes, int n_in,
                              void* d_out, int out_size, void* d_ws, size_t ws_size,
                              hipStream_t stream) {
    (void)in_sizes; (void)n_in; (void)out_size; (void)ws_size;

    const float* x     = (const float*)d_in[0];
    const int*   eidx  = (const int*)d_in[1];
    const int*   batch = (const int*)d_in[2];
    const float* gfeat = (const float*)d_in[3];
    const float* Wl1 = (const float*)d_in[4];  const float* bl1 = (const float*)d_in[5];
    const float* Wr1 = (const float*)d_in[6];  const float* br1 = (const float*)d_in[7];
    const float* att1 = (const float*)d_in[8]; const float* bias1 = (const float*)d_in[9];
    const float* g1 = (const float*)d_in[10];  const float* be1 = (const float*)d_in[11];
    const float* m1 = (const float*)d_in[12];  const float* v1 = (const float*)d_in[13];
    const float* Wl2 = (const float*)d_in[14]; const float* bl2 = (const float*)d_in[15];
    const float* Wr2 = (const float*)d_in[16]; const float* br2 = (const float*)d_in[17];
    const float* att2 = (const float*)d_in[18]; const float* bias2 = (const float*)d_in[19];
    const float* g2 = (const float*)d_in[20];  const float* be2 = (const float*)d_in[21];
    const float* m2 = (const float*)d_in[22];  const float* v2 = (const float*)d_in[23];
    const float* fc1W = (const float*)d_in[24]; const float* fc1b = (const float*)d_in[25];
    const float* fc2W = (const float*)d_in[26]; const float* fc2b = (const float*)d_in[27];
    float* out = (float*)d_out;

    const int* srcIdx = eidx;
    const int* dstIdx = eidx + E_EDGES;

    // ---- workspace layout (phase-reused regions)
    char* ws = (char*)d_ws;
    size_t off = 0;
    auto take = [&](size_t bytes) -> char* {
        char* p = ws + off;
        off += (bytes + 255) & ~(size_t)255;
        return p;
    };
    float* R1 = (float*)take((size_t)N_NODES * D2 * 4);   // L1: [xl1|xr1]  L2: xl2
    float* R2 = (float*)take((size_t)N_NODES * D2 * 4);   // L1: agg1       L2: xr2
    float* R3 = (float*)take((size_t)N_NODES * D2 * 4);   // L2: agg2 / h2
    __bf16* H1B = (__bf16*)take((size_t)N_NODES * D1 * 2);
    float* SC = (float*)take((size_t)(E_EDGES + N_NODES) * HEADS * 4);
    float* MB = (float*)take((size_t)N_NODES * HEADS * 4);
    float* DN = (float*)take((size_t)N_NODES * HEADS * 4);
    __bf16* WTl = (__bf16*)take((size_t)D1 * D2 * 2);
    __bf16* WTr = (__bf16*)take((size_t)D1 * D2 * 2);
    float* PL = (float*)take((size_t)G_GRAPHS * D2 * 4);
    float* CT = (float*)take((size_t)G_GRAPHS * 4);

    float* xl1 = R1;
    float* xr1 = R1 + (size_t)N_NODES * D1;
    float* agg1 = R2;
    float* xl2 = R1;
    float* xr2 = R2;
    float* agg2 = R3;

    const int TB = 256;
    auto blocks   = [](int n, int tb) { return (n + tb - 1) / tb; };
    auto blocksLL = [](long long n, int tb) { return (int)((n + tb - 1) / tb); };
    const int EEH = (E_EDGES + N_NODES) * HEADS;
    const long long E1 = (long long)(E_EDGES + N_NODES) * D1;
    const long long E2 = (long long)(E_EDGES + N_NODES) * D2;
    const float NEG_INF = -__builtin_huge_valf();

    // ================= layer 1 =================
    fill_f32<<<blocks(N_NODES * D1, TB), TB, 0, stream>>>(agg1, 0.0f, N_NODES * D1);
    fill_f32<<<blocks(N_NODES * HEADS, TB), TB, 0, stream>>>(MB, NEG_INF, N_NODES * HEADS);
    fill_f32<<<blocks(N_NODES * HEADS, TB), TB, 0, stream>>>(DN, 0.0f, N_NODES * HEADS);

    l1_transform<<<blocks(N_NODES * D1, TB), TB, 0, stream>>>(x, Wl1, bl1, Wr1, br1, xl1, xr1);
    edge_score_wave<C1, D1><<<blocksLL((long long)EEH * 32, TB), TB, 0, stream>>>(
        xl1, xr1, att1, srcIdx, dstIdx, SC, MB);
    edge_exp<<<blocks(EEH, TB), TB, 0, stream>>>(srcIdx, dstIdx, SC, MB, DN);
    edge_scatter<C1, D1><<<blocksLL(E1, TB), TB, 0, stream>>>(xl1, srcIdx, dstIdx, SC, DN, agg1);
    bn_relu<<<blocks(N_NODES * D1, TB), TB, 0, stream>>>(agg1, bias1, g1, be1, m1, v1,
                                                         nullptr, H1B, D1, N_NODES * D1);

    // ================= layer 2 GEMMs (dual bf16 WMMA, LDS-staged B) =========
    transpose_cvt<<<blocks(D1 * D2, TB), TB, 0, stream>>>(Wl2, WTl, D1, D2);
    transpose_cvt<<<blocks(D1 * D2, TB), TB, 0, stream>>>(Wr2, WTr, D1, D2);

    dim3 gGrid(N_NODES / 64, D2 / 64);   // (2048, 8); 4 waves/block; 16x64x2 per wave
    wmma_gemm_dual<<<gGrid, 128, 0, stream>>>(H1B, WTl, WTr, bl2, br2, xl2, xr2);

    // ================= layer 2 attention =================
    fill_f32<<<blocks(N_NODES * HEADS, TB), TB, 0, stream>>>(MB, NEG_INF, N_NODES * HEADS);
    fill_f32<<<blocks(N_NODES * HEADS, TB), TB, 0, stream>>>(DN, 0.0f, N_NODES * HEADS);
    fill_f32<<<blocks(N_NODES * D2, TB), TB, 0, stream>>>(agg2, 0.0f, N_NODES * D2);

    edge_score_wave<C2, D2><<<blocksLL((long long)EEH * 32, TB), TB, 0, stream>>>(
        xl2, xr2, att2, srcIdx, dstIdx, SC, MB);
    edge_exp<<<blocks(EEH, TB), TB, 0, stream>>>(srcIdx, dstIdx, SC, MB, DN);
    edge_scatter<C2, D2><<<blocksLL(E2, TB), TB, 0, stream>>>(xl2, srcIdx, dstIdx, SC, DN, agg2);
    bn_relu<<<blocks(N_NODES * D2, TB), TB, 0, stream>>>(agg2, bias2, g2, be2, m2, v2,
                                                         agg2, nullptr, D2, N_NODES * D2);

    // ================= pooling + head =================
    fill_f32<<<blocks(G_GRAPHS * D2, TB), TB, 0, stream>>>(PL, 0.0f, G_GRAPHS * D2);
    fill_f32<<<blocks(G_GRAPHS, TB), TB, 0, stream>>>(CT, 0.0f, G_GRAPHS);
    pool_sum<<<blocks(N_NODES * D2, TB), TB, 0, stream>>>(agg2, batch, PL);
    pool_count<<<blocks(N_NODES, TB), TB, 0, stream>>>(batch, CT);
    head_kernel<<<G_GRAPHS, 128, 0, stream>>>(PL, CT, gfeat, fc1W, fc1b, fc2W, fc2b, out);
}
